// Net_38998303047925
// MI455X (gfx1250) — matrix operand order repaired and checked by reading, not compiled, not run on hardware
//
#include <hip/hip_runtime.h>
#include <hip/hip_bf16.h>

typedef __attribute__((ext_vector_type(16))) _Float16 v16h;
typedef __attribute__((ext_vector_type(8)))  float    v8f;
typedef __attribute__((ext_vector_type(4)))  float    f4;

// ---------------- helpers ----------------
__device__ __forceinline__ float leakyf(float v, float s) { return v > 0.f ? v : s * v; }

// order-preserving float<->uint encoding for atomicMax on floats
__device__ __forceinline__ unsigned ford(float f) {
    unsigned u = __float_as_uint(f);
    return (u & 0x80000000u) ? ~u : (u | 0x80000000u);
}
__device__ __forceinline__ float ford_inv(unsigned u) {
    return __uint_as_float((u & 0x80000000u) ? (u & 0x7FFFFFFFu) : ~u);
}
#define FORD_NEG_INF 0x007FFFFFu  // ford(-inf) precomputed for host use

// build an f16 A-fragment (16 elems) from four float4 loads
__device__ __forceinline__ v16h cvt_frag(f4 x0, f4 x1, f4 x2, f4 x3) {
    v16h a;
    #pragma unroll
    for (int j = 0; j < 4; ++j) {
        a[j]      = (_Float16)x0[j];
        a[j + 4]  = (_Float16)x1[j];
        a[j + 8]  = (_Float16)x2[j];
        a[j + 12] = (_Float16)x3[j];
    }
    return a;
}

// ---------------- elementwise / scatter kernels ----------------
__global__ void k_fill_f32(float* __restrict__ p, float v, int n) {
    int i = blockIdx.x * blockDim.x + threadIdx.x;
    if (i < n) p[i] = v;
}
__global__ void k_fill_u32(unsigned* __restrict__ p, unsigned v, int n) {
    int i = blockIdx.x * blockDim.x + threadIdx.x;
    if (i < n) p[i] = v;
}
__global__ void k_count(const int* __restrict__ idx, float* __restrict__ cnt, int n) {
    int i = blockIdx.x * blockDim.x + threadIdx.x;
    if (i < n) atomicAdd(&cnt[idx[i]], 1.0f);
}
__global__ void k_div_rows64(float* __restrict__ m, const float* __restrict__ cnt, int rows) {
    int i = blockIdx.x * blockDim.x + threadIdx.x;
    if (i < rows * 64) m[i] /= fmaxf(cnt[i >> 6], 1.0f);
}
__global__ void k_add(float* __restrict__ a, const float* __restrict__ b, int n) {
    int i = blockIdx.x * blockDim.x + threadIdx.x;
    if (i < n) a[i] += b[i];
}
__global__ void k_add_bias64(float* __restrict__ a, const float* __restrict__ b, int rows) {
    int i = blockIdx.x * blockDim.x + threadIdx.x;
    if (i < rows * 64) a[i] += b[i & 63];
}
// pack W[K,Nout] (f32, row-major) -> Wt[Nout][Kpad] (f16, transposed, zero-padded)
__global__ void k_pack_w(const float* __restrict__ W, int K, int Nout,
                         _Float16* __restrict__ Wt, int Kpad) {
    int i = blockIdx.x * blockDim.x + threadIdx.x;
    if (i >= Nout * Kpad) return;
    int n = i / Kpad, k = i - n * Kpad;
    Wt[i] = (k < K) ? (_Float16)W[(size_t)k * Nout + n] : (_Float16)0.f;
}
// cat = concat(out[N,64], x[N,2]) -> [N,96] zero-padded
__global__ void k_concat_out_x(const float* __restrict__ o, const float* __restrict__ x,
                               float* __restrict__ cat, int N) {
    int n = blockIdx.x * blockDim.x + threadIdx.x;
    if (n >= N) return;
    float* d = cat + (size_t)n * 96;
    const float* s = o + (size_t)n * 64;
    #pragma unroll
    for (int k = 0; k < 64; ++k) d[k] = s[k];
    d[64] = x[2 * n]; d[65] = x[2 * n + 1];
    #pragma unroll
    for (int k = 66; k < 96; ++k) d[k] = 0.f;
}
// dst[N,128] = concat(agg[N,64], u[batch[n]])
__global__ void k_build_nodeu(const float* __restrict__ a64, const float* __restrict__ u,
                              const int* __restrict__ batch, float* __restrict__ dst, int N) {
    int i = blockIdx.x * blockDim.x + threadIdx.x;
    if (i >= N * 128) return;
    int n = i >> 7, c = i & 127;
    dst[i] = (c < 64) ? a64[(size_t)n * 64 + c] : u[(size_t)batch[n] * 64 + (c - 64)];
}
// dst[B,128] = concat(u[B,64], gmean[B,64])
__global__ void k_build_ucat(const float* __restrict__ u, const float* __restrict__ g,
                             float* __restrict__ dst, int B) {
    int i = blockIdx.x * blockDim.x + threadIdx.x;
    if (i >= B * 128) return;
    int b = i >> 7, c = i & 127;
    dst[i] = (c < 64) ? u[b * 64 + c] : g[b * 64 + (c - 64)];
}
__global__ void k_node_graph_sum(const float* __restrict__ xv, const int* __restrict__ batch,
                                 float* __restrict__ gsum, int N) {
    int i = blockIdx.x * blockDim.x + threadIdx.x;
    if (i >= N * 64) return;
    atomicAdd(&gsum[(size_t)batch[i >> 6] * 64 + (i & 63)], xv[i]);
}
__global__ void k_matvec64(const float* __restrict__ H, const float* __restrict__ v,
                           float* __restrict__ o, int N) {
    int n = blockIdx.x * blockDim.x + threadIdx.x;
    if (n >= N) return;
    const float* r = H + (size_t)n * 64;
    float s = 0.f;
    #pragma unroll
    for (int k = 0; k < 64; ++k) s += r[k] * v[k];
    o[n] = s;
}

// ---------------- GAT edge-softmax kernels (edges 0..E-1, self-loops E..E+N-1) ----------------
__global__ void k_gat_max(const float* __restrict__ als, const float* __restrict__ ald,
                          const int* __restrict__ src, const int* __restrict__ dst,
                          int E, int N, unsigned* __restrict__ mord) {
    int t = blockIdx.x * blockDim.x + threadIdx.x;
    if (t >= E + N) return;
    int s, d;
    if (t < E) { s = src[t]; d = dst[t]; } else { s = d = t - E; }
    float a = leakyf(als[s] + ald[d], 0.2f);
    atomicMax(&mord[d], ford(a));
}
__global__ void k_gat_denom(const float* __restrict__ als, const float* __restrict__ ald,
                            const int* __restrict__ src, const int* __restrict__ dst,
                            int E, int N, const unsigned* __restrict__ mord,
                            float* __restrict__ denom) {
    int t = blockIdx.x * blockDim.x + threadIdx.x;
    if (t >= E + N) return;
    int s, d;
    if (t < E) { s = src[t]; d = dst[t]; } else { s = d = t - E; }
    float a = leakyf(als[s] + ald[d], 0.2f);
    atomicAdd(&denom[d], expf(a - ford_inv(mord[d])));
}
__global__ void k_gat_scatter(const float* __restrict__ h,
                              const float* __restrict__ als, const float* __restrict__ ald,
                              const int* __restrict__ src, const int* __restrict__ dst,
                              int E, int N, const unsigned* __restrict__ mord,
                              const float* __restrict__ denom, float* __restrict__ out) {
    int t = blockIdx.x * blockDim.x + threadIdx.x;
    if (t >= E + N) return;
    int s, d;
    if (t < E) { s = src[t]; d = dst[t]; } else { s = d = t - E; }
    float a = leakyf(als[s] + ald[d], 0.2f);
    float coef = expf(a - ford_inv(mord[d])) / denom[d];
    const float* hs = h + (size_t)s * 64;
    float* o = out + (size_t)d * 64;
    #pragma unroll 4
    for (int c = 0; c < 64; ++c) atomicAdd(&o[c], coef * hs[c]);
}

// ---------------- generic WMMA GEMM: Out[M,Nout] = A[M,K] @ Wt^T + bias ----------------
// K is pre-padded (K%32==0, lda==K). Wt is [Nout][K] f16 (transposed, zero-padded).
// One wave32 per 16x16 D tile. Out-of-range rows/cols are CLAMPED on load (their
// products land only in D elements the guarded store never writes), so the inner
// loop is branch-free: b128 loads + cvt + v_wmma_f32_16x16x32_f16.
__global__ __launch_bounds__(32) void
k_gemm_wmma(const float* __restrict__ A, int M, int K,
            const _Float16* __restrict__ Wt, int Nout,
            const float* __restrict__ bias, int act,
            float* __restrict__ Out) {
    const int lane = threadIdx.x & 31;
    const int hh = lane >> 4;
    const int lm = lane & 15;
    const int m = blockIdx.x * 16 + lm;
    const int n = blockIdx.y * 16 + lm;
    const int mc = m < M ? m : M - 1;
    const int nc = n < Nout ? n : Nout - 1;

    const float* ap = A + (size_t)mc * K + hh * 16;
    const _Float16* wp = Wt + (size_t)nc * K + hh * 16;

    v8f c = {};
    for (int kb = 0; kb < K; kb += 32) {
        f4 x0 = *(const f4*)(ap + kb);
        f4 x1 = *(const f4*)(ap + kb + 4);
        f4 x2 = *(const f4*)(ap + kb + 8);
        f4 x3 = *(const f4*)(ap + kb + 12);
        v16h a = cvt_frag(x0, x1, x2, x3);
        v16h b = *(const v16h*)(wp + kb);
        c = __builtin_amdgcn_wmma_f32_16x16x32_f16(false, a, false, b, (short)0, c, false, false);
    }
    const float bb = (bias != nullptr && n < Nout) ? bias[n] : 0.f;
    #pragma unroll
    for (int j = 0; j < 8; ++j) {
        int row = blockIdx.x * 16 + hh * 8 + j;
        if (row < M && n < Nout) {
            float v = c[j] + bb;
            if (act == 1) v = leakyf(v, 0.01f);
            Out[(size_t)row * Nout + n] = v;
        }
    }
}

// ---------------- fused edge-message WMMA, Cn=64 (K=68 -> Kpad=96) ----------------
// msg[e] = [feat[col[e]](64), eattr[e](4)] @ W + b ; atomicAdd into aggsum[row[e]]
__global__ __launch_bounds__(32) void
k_edge_msg64(const float* __restrict__ feat, const float* __restrict__ eattr,
             const int* __restrict__ colv, const int* __restrict__ rowv,
             const _Float16* __restrict__ Wt /*[64][96]*/, const float* __restrict__ bias,
             float* __restrict__ aggsum, int E) {
    const int lane = threadIdx.x & 31;
    const int hh = lane >> 4;
    const int lm = lane & 15;
    const int e = blockIdx.x * 16 + lm;
    const int n = blockIdx.y * 16 + lm;
    const int ec = e < E ? e : E - 1;            // clamp: only unstored D rows polluted
    const int ci = colv[ec];

    const float* fp = feat + (size_t)ci * 64 + hh * 16;
    const _Float16* wp = Wt + (size_t)n * 96 + hh * 16;

    v8f c = {};
    #pragma unroll
    for (int kb = 0; kb < 64; kb += 32) {        // node-feature k-steps
        f4 x0 = *(const f4*)(fp + kb);
        f4 x1 = *(const f4*)(fp + kb + 4);
        f4 x2 = *(const f4*)(fp + kb + 8);
        f4 x3 = *(const f4*)(fp + kb + 12);
        v16h a = cvt_frag(x0, x1, x2, x3);
        v16h b = *(const v16h*)(wp + kb);
        c = __builtin_amdgcn_wmma_f32_16x16x32_f16(false, a, false, b, (short)0, c, false, false);
    }
    // k-step 2: k=64..67 hold eattr (lanes hh==0, elems 0..3), rest zero
    {
        f4 ea = *(const f4*)(eattr + (size_t)ec * 4);
        const bool lo = (hh == 0);
        v16h a;
        #pragma unroll
        for (int j = 0; j < 16; ++j) a[j] = (_Float16)0.f;
        a[0] = lo ? (_Float16)ea[0] : (_Float16)0.f;
        a[1] = lo ? (_Float16)ea[1] : (_Float16)0.f;
        a[2] = lo ? (_Float16)ea[2] : (_Float16)0.f;
        a[3] = lo ? (_Float16)ea[3] : (_Float16)0.f;
        v16h b = *(const v16h*)(wp + 64);
        c = __builtin_amdgcn_wmma_f32_16x16x32_f16(false, a, false, b, (short)0, c, false, false);
    }
    const float bb = bias[n];
    #pragma unroll
    for (int j = 0; j < 8; ++j) {
        int e2 = blockIdx.x * 16 + hh * 8 + j;
        if (e2 < E) {
            int r = rowv[e2];
            atomicAdd(&aggsum[(size_t)r * 64 + n], c[j] + bb);
        }
    }
}

// ---------------- fused edge-message WMMA, Cn=2 (conv1, K=6 -> Kpad=32) ----------------
__global__ __launch_bounds__(32) void
k_edge_msg2(const float* __restrict__ feat /*[N,2]*/, const float* __restrict__ eattr,
            const int* __restrict__ colv, const int* __restrict__ rowv,
            const _Float16* __restrict__ Wt /*[64][32]*/, const float* __restrict__ bias,
            float* __restrict__ aggsum, int E) {
    const int lane = threadIdx.x & 31;
    const int hh = lane >> 4;
    const int lm = lane & 15;
    const int e = blockIdx.x * 16 + lm;
    const int n = blockIdx.y * 16 + lm;
    const int ec = e < E ? e : E - 1;
    const int ci = colv[ec];

    const float xf0 = feat[(size_t)ci * 2];
    const float xf1 = feat[(size_t)ci * 2 + 1];
    f4 ea = *(const f4*)(eattr + (size_t)ec * 4);
    const bool lo = (hh == 0);

    v16h a;
    #pragma unroll
    for (int j = 0; j < 16; ++j) a[j] = (_Float16)0.f;
    a[0] = lo ? (_Float16)xf0   : (_Float16)0.f;   // k=0
    a[1] = lo ? (_Float16)xf1   : (_Float16)0.f;   // k=1
    a[2] = lo ? (_Float16)ea[0] : (_Float16)0.f;   // k=2
    a[3] = lo ? (_Float16)ea[1] : (_Float16)0.f;   // k=3
    a[4] = lo ? (_Float16)ea[2] : (_Float16)0.f;   // k=4
    a[5] = lo ? (_Float16)ea[3] : (_Float16)0.f;   // k=5
    v16h b = *(const v16h*)(Wt + (size_t)n * 32 + hh * 16);

    v8f c = {};
    c = __builtin_amdgcn_wmma_f32_16x16x32_f16(false, a, false, b, (short)0, c, false, false);

    const float bb = bias[n];
    #pragma unroll
    for (int j = 0; j < 8; ++j) {
        int e2 = blockIdx.x * 16 + hh * 8 + j;
        if (e2 < E) {
            int r = rowv[e2];
            atomicAdd(&aggsum[(size_t)r * 64 + n], c[j] + bb);
        }
    }
}

// ---------------- host orchestration ----------------
extern "C" void kernel_launch(void* const* d_in, const int* in_sizes, int n_in,
                              void* d_out, int out_size, void* d_ws, size_t ws_size,
                              hipStream_t stream) {
    (void)n_in; (void)ws_size;
    const float* x      = (const float*)d_in[0];
    const int*   eidx   = (const int*)d_in[1];
    const float* eattr  = (const float*)d_in[2];
    const int*   batch  = (const int*)d_in[3];
    const float* c1m1W  = (const float*)d_in[5];
    const float* c1m1b  = (const float*)d_in[6];
    const float* c1m2W  = (const float*)d_in[7];
    const float* c1m2b  = (const float*)d_in[8];
    const float* c1gW   = (const float*)d_in[9];
    const float* c1gb   = (const float*)d_in[10];
    const float* m1W    = (const float*)d_in[11];
    const float* m1b    = (const float*)d_in[12];
    const float* m2W    = (const float*)d_in[13];
    const float* m2b    = (const float*)d_in[14];
    const float* gW     = (const float*)d_in[15];
    const float* gb     = (const float*)d_in[16];
    const float* gatW   = (const float*)d_in[17];
    const float* gatAs  = (const float*)d_in[18];
    const float* gatAd  = (const float*)d_in[19];
    const float* gatB   = (const float*)d_in[20];
    const float* o1W    = (const float*)d_in[21];
    const float* o1b    = (const float*)d_in[22];
    const float* o2W    = (const float*)d_in[23];
    const float* o2b    = (const float*)d_in[24];
    const float* o3W    = (const float*)d_in[25];
    const float* o3b    = (const float*)d_in[26];

    const int N = in_sizes[3];
    const int E = in_sizes[1] / 2;
    const int B = out_size;            // num_graphs
    const int* rowv = eidx;            // edge_index[0]
    const int* colv = eidx + E;        // edge_index[1]

    // workspace carve-up (floats, each chunk 64B aligned)
    float* ws = (float*)d_ws;
    size_t off = 0;
    auto alloc = [&](size_t nf) { float* p = ws + off; off += (nf + 15) & ~(size_t)15; return p; };
    float* deg    = alloc(N);
    float* bcnt   = alloc(B);
    float* outbuf = alloc((size_t)N * 64);
    float* glob   = alloc((size_t)B * 64);
    float* utmp   = alloc((size_t)B * 64);
    float* vs     = alloc((size_t)N * 64);
    float* agg    = alloc((size_t)N * 64);
    float* xin    = alloc((size_t)N * 128);
    float* cat    = alloc((size_t)N * 96);
    float* hbuf   = alloc((size_t)N * 64);
    float* als    = alloc(N);
    float* ald    = alloc(N);
    unsigned* mord = (unsigned*)alloc(N);
    float* denom  = alloc(N);
    float* gout   = alloc((size_t)N * 64);
    float* gsum   = alloc((size_t)B * 64);
    float* ucat   = alloc((size_t)B * 128);
    float* hd1    = alloc((size_t)B * 64);
    float* hd2    = alloc((size_t)B * 32);

    const dim3 blk256(256);
    auto g1 = [](int n) { return dim3((unsigned)((n + 255) / 256)); };
    const dim3 wblk(32);

    // pack weights -> transposed zero-padded f16 [Nout][Kpad]
    auto pack = [&](const float* W, int K, int Nout, int Kpad) -> _Float16* {
        _Float16* p = (_Float16*)alloc(((size_t)Kpad * Nout + 1) / 2);
        k_pack_w<<<g1(Kpad * Nout), blk256, 0, stream>>>(W, K, Nout, p, Kpad);
        return p;
    };
    _Float16* Wt_c1m1 = pack(c1m1W, 6, 64, 32);
    _Float16* Wt_c1m2 = pack(c1m2W, 128, 64, 128);
    _Float16* Wt_c1g  = pack(c1gW, 128, 64, 128);
    _Float16 *Wt_m1[4], *Wt_m2[4], *Wt_g[4], *Wt_gat[4];
    for (int i = 0; i < 4; ++i) {
        Wt_m1[i]  = pack(m1W + (size_t)i * 68 * 64, 68, 64, 96);
        Wt_m2[i]  = pack(m2W + (size_t)i * 128 * 64, 128, 64, 128);
        Wt_g[i]   = pack(gW + (size_t)i * 128 * 64, 128, 64, 128);
        Wt_gat[i] = pack(gatW + (size_t)i * 66 * 64, 66, 64, 96);
    }
    _Float16* Wt_o1 = pack(o1W, 64, 64, 64);
    _Float16* Wt_o2 = pack(o2W, 64, 32, 64);
    _Float16* Wt_o3 = pack(o3W, 32, 1, 32);

    auto gemm = [&](const float* A, int M, int Kpad, const _Float16* Wt, int Nout,
                    const float* bias, int act, float* Out) {
        dim3 grid((unsigned)((M + 15) / 16), (unsigned)((Nout + 15) / 16));
        k_gemm_wmma<<<grid, wblk, 0, stream>>>(A, M, Kpad, Wt, Nout, bias, act, Out);
    };
    const dim3 egrid((unsigned)((E + 15) / 16), 4);
    auto graph_mean = [&](const float* xv) {
        k_fill_f32<<<g1(B * 64), blk256, 0, stream>>>(gsum, 0.f, B * 64);
        k_node_graph_sum<<<g1(N * 64), blk256, 0, stream>>>(xv, batch, gsum, N);
        k_div_rows64<<<g1(B * 64), blk256, 0, stream>>>(gsum, bcnt, B);
    };

    // fixed degree / graph-size counts
    k_fill_f32<<<g1(N), blk256, 0, stream>>>(deg, 0.f, N);
    k_count<<<g1(E), blk256, 0, stream>>>(rowv, deg, E);
    k_fill_f32<<<g1(B), blk256, 0, stream>>>(bcnt, 0.f, B);
    k_count<<<g1(N), blk256, 0, stream>>>(batch, bcnt, N);

    // -------- conv1 = MyLayer(x[N,2], u0=0) --------
    k_fill_f32<<<g1(N * 64), blk256, 0, stream>>>(agg, 0.f, N * 64);
    k_edge_msg2<<<egrid, wblk, 0, stream>>>(x, eattr, colv, rowv, Wt_c1m1, c1m1b, agg, E);
    k_div_rows64<<<g1(N * 64), blk256, 0, stream>>>(agg, deg, N);
    k_fill_f32<<<g1(B * 64), blk256, 0, stream>>>(glob, 0.f, B * 64); // u0
    k_build_nodeu<<<g1(N * 128), blk256, 0, stream>>>(agg, glob, batch, xin, N);
    gemm(xin, N, 128, Wt_c1m2, 64, c1m2b, 0, outbuf);               // out = x_new
    graph_mean(outbuf);
    k_build_ucat<<<g1(B * 128), blk256, 0, stream>>>(glob, gsum, ucat, B);
    gemm(ucat, B, 128, Wt_c1g, 64, c1gb, 0, glob);                  // glob = u_new

    // -------- 4 stacked (GATConv + MyLayer) --------
    const int T = E + N;  // edges + self-loops
    for (int i = 0; i < 4; ++i) {
        const float* as_i = gatAs + (size_t)i * 64;
        const float* ad_i = gatAd + (size_t)i * 64;
        const float* gb_i = gatB + (size_t)i * 64;

        // GATConv on concat(out, x) [N,96-padded]
        k_concat_out_x<<<g1(N), blk256, 0, stream>>>(outbuf, x, cat, N);
        gemm(cat, N, 96, Wt_gat[i], 64, nullptr, 0, hbuf);          // h = cat @ W
        k_matvec64<<<g1(N), blk256, 0, stream>>>(hbuf, as_i, als, N);
        k_matvec64<<<g1(N), blk256, 0, stream>>>(hbuf, ad_i, ald, N);
        k_fill_u32<<<g1(N), blk256, 0, stream>>>(mord, FORD_NEG_INF, N);
        k_gat_max<<<g1(T), blk256, 0, stream>>>(als, ald, rowv, colv, E, N, mord);
        k_fill_f32<<<g1(N), blk256, 0, stream>>>(denom, 0.f, N);
        k_gat_denom<<<g1(T), blk256, 0, stream>>>(als, ald, rowv, colv, E, N, mord, denom);
        k_fill_f32<<<g1(N * 64), blk256, 0, stream>>>(gout, 0.f, N * 64);
        k_gat_scatter<<<g1(T), blk256, 0, stream>>>(hbuf, als, ald, rowv, colv, E, N, mord, denom, gout);
        k_add_bias64<<<g1(N * 64), blk256, 0, stream>>>(gout, gb_i, N);

        // MyLayer on gat output [N,64]
        k_fill_f32<<<g1(N * 64), blk256, 0, stream>>>(agg, 0.f, N * 64);
        k_edge_msg64<<<egrid, wblk, 0, stream>>>(gout, eattr, colv, rowv, Wt_m1[i],
                                                 m1b + (size_t)i * 64, agg, E);
        k_div_rows64<<<g1(N * 64), blk256, 0, stream>>>(agg, deg, N);
        k_build_nodeu<<<g1(N * 128), blk256, 0, stream>>>(agg, glob, batch, xin, N);
        gemm(xin, N, 128, Wt_m2[i], 64, m2b + (size_t)i * 64, 0, vs); // vs = x_new
        graph_mean(vs);
        k_build_ucat<<<g1(B * 128), blk256, 0, stream>>>(glob, gsum, ucat, B);
        gemm(ucat, B, 128, Wt_g[i], 64, gb + (size_t)i * 64, 0, utmp); // g = u_new
        k_add<<<g1(N * 64), blk256, 0, stream>>>(outbuf, vs, N * 64);
        k_add<<<g1(B * 64), blk256, 0, stream>>>(glob, utmp, B * 64);
    }

    // -------- output head on glob [B,64] --------
    gemm(glob, B, 64, Wt_o1, 64, o1b, 1, hd1);
    gemm(hd1, B, 64, Wt_o2, 32, o2b, 1, hd2);
    gemm(hd2, B, 32, Wt_o3, 1, o3b, 0, (float*)d_out);
}
